// Attention_16518444220655
// MI455X (gfx1250) — compile-verified, matrix operand order checked by reading, not run on hardware
//
#include <hip/hip_runtime.h>
#include <hip/hip_bf16.h>
#include <math.h>
#include <stdint.h>

#define SEQ      2048
#define DIM      4096
#define NHEADS   32
#define NKVHEADS 8
#define HEADDIM  128
#define QDIM     (NHEADS * HEADDIM)    // 4096
#define KVDIM    (NKVHEADS * HEADDIM)  // 1024
#define SCALE_F  0.08838834764831845f  // 128^-0.5

typedef __bf16 bf16x16 __attribute__((ext_vector_type(16)));
typedef __bf16 bf16x8  __attribute__((ext_vector_type(8)));
typedef float  floatx8 __attribute__((ext_vector_type(8)));

// Force global (addrspace 1) loads so hot loops emit global_load_b128 and
// never flat_load_b128 (FLAT ticks DScnt and steals LDS-path issue slots).
__device__ __forceinline__ bf16x8 gload8(const __bf16* p) {
  return *(const __attribute__((address_space(1))) bf16x8*)(uintptr_t)p;
}
__device__ __forceinline__ bf16x16 gload16(const __bf16* p) {
  return *(const __attribute__((address_space(1))) bf16x16*)(uintptr_t)p;
}

// ---------------------------------------------------------------------------
// 1) f32 -> bf16 row-major convert
// ---------------------------------------------------------------------------
__global__ __launch_bounds__(256) void convert_bf16_kernel(
    const float* __restrict__ src, __bf16* __restrict__ dst, int n) {
  int tid = blockIdx.x * 256 + threadIdx.x;
  if (tid < n) dst[tid] = (__bf16)src[tid];
}

// ---------------------------------------------------------------------------
// 2) Pack an f32 matrix [Ktiles*32 x Ntiles*16] (leading dim ld) into the
//    WMMA bf16 B-operand fragment layout:
//      dst[((nt*Ktiles + kt)*32 + lane)*16 + e]
//    where element e of lane l maps to K-offset:
//      e<8  : K = e      + 8*(l>=16)
//      e>=8 : K = e + 8  + 8*(l>=16)
//    and N-offset = l & 15.  (ISA 05_wmma 16-bit operand layout.)
// ---------------------------------------------------------------------------
__global__ __launch_bounds__(256) void pack_b_kernel(
    const float* __restrict__ src, int ld, int Ktiles, int Ntiles,
    __bf16* __restrict__ dst, int total) {
  int tid = blockIdx.x * 256 + threadIdx.x;
  if (tid >= total) return;
  int e    = tid & 15;
  int lane = (tid >> 4) & 31;
  int t    = tid >> 9;
  int kt   = t % Ktiles;
  int nt   = t / Ktiles;
  int lh   = (lane >> 4) << 3;               // 0 or 8
  int K    = (e < 8) ? (e + lh) : (e + 8 + lh);
  int row  = kt * 32 + K;
  int col  = nt * 16 + (lane & 15);
  dst[tid] = (__bf16)src[(size_t)row * ld + col];
}

// ---------------------------------------------------------------------------
// 3) Register-blocked WMMA GEMM: C[M x N] = A[M x K] * B[K x N]
//    A: bf16 row-major (lda), B: packed fragment layout, C: f32 row-major.
//    One wave computes a 64x32 macro-tile (4 M-tiles x 2 N-tiles) -> each
//    loaded fragment feeds multiple WMMAs (A reused 2x, B reused 4x):
//    per K=32 step, 6 KB moved for 128 KFLOP (~21 FLOP/byte).
//    8 waves per block.
// ---------------------------------------------------------------------------
#define GMT 4   // M-tiles per wave (64 rows)
#define GNT 2   // N-tiles per wave (32 cols)

__global__ __launch_bounds__(256) void gemm_wmma_bf16_kernel(
    const __bf16* __restrict__ A, int lda,
    const __bf16* __restrict__ Bp,
    float* __restrict__ C, int ldc,
    int NMacro /* N/32 */, int Ktiles) {
  int lane = threadIdx.x & 31;
  int wave = threadIdx.x >> 5;
  int wid  = blockIdx.x * 8 + wave;
  int tn   = wid % NMacro;                   // macro column (32 cols)
  int tm   = wid / NMacro;                   // macro row    (64 rows)

  int l16  = lane & 15;
  int kofs = (lane >> 4) << 3;               // 0 or 8

  const __bf16* arow[GMT];
#pragma unroll
  for (int mt = 0; mt < GMT; ++mt)
    arow[mt] = A + (size_t)(tm * 64 + mt * 16 + l16) * lda + kofs;

  const size_t bstride_nt = (size_t)Ktiles * 512;  // elements per n-tile
  const __bf16* bptr = Bp + ((size_t)(tn * GNT) * Ktiles * 32 + lane) * 16;

  floatx8 acc[GMT][GNT];
#pragma unroll
  for (int mt = 0; mt < GMT; ++mt)
#pragma unroll
    for (int nt = 0; nt < GNT; ++nt) acc[mt][nt] = (floatx8){};

  for (int kt = 0; kt < Ktiles; ++kt) {
    // prefetch the two contiguous packed-B streams a few steps ahead
    __builtin_prefetch((const void*)(bptr + 8 * 512), 0, 0);
    __builtin_prefetch((const void*)(bptr + bstride_nt + 8 * 512), 0, 0);

    bf16x16 av[GMT];
#pragma unroll
    for (int mt = 0; mt < GMT; ++mt) {
      bf16x8 a0 = gload8(arow[mt]);
      bf16x8 a1 = gload8(arow[mt] + 16);
#pragma unroll
      for (int i = 0; i < 8; ++i) { av[mt][i] = a0[i]; av[mt][i + 8] = a1[i]; }
    }
    bf16x16 bv[GNT];
#pragma unroll
    for (int nt = 0; nt < GNT; ++nt)
      bv[nt] = gload16(bptr + (size_t)nt * bstride_nt);

#pragma unroll
    for (int mt = 0; mt < GMT; ++mt)
#pragma unroll
      for (int nt = 0; nt < GNT; ++nt)
        acc[mt][nt] = __builtin_amdgcn_wmma_f32_16x16x32_bf16(
            false, av[mt], false, bv[nt], (short)0, acc[mt][nt], false, false);

#pragma unroll
    for (int mt = 0; mt < GMT; ++mt) arow[mt] += 32;
    bptr += 512;
  }

#pragma unroll
  for (int mt = 0; mt < GMT; ++mt) {
#pragma unroll
    for (int nt = 0; nt < GNT; ++nt) {
      int ccol = tn * 32 + nt * 16 + l16;
#pragma unroll
      for (int r = 0; r < 8; ++r) {
        int crow = tm * 64 + mt * 16 + r + kofs;  // M = r + 8*(lane>=16)
        C[(size_t)crow * ldc + ccol] = acc[mt][nt][r];
      }
    }
  }
}

// ---------------------------------------------------------------------------
// 4) RoPE: f32 [S x nheads*128] -> bf16 [S x nheads*128]
// ---------------------------------------------------------------------------
__global__ __launch_bounds__(256) void rope_kernel(
    const float* __restrict__ src,
    const float* __restrict__ cosb, const float* __restrict__ sinb,
    __bf16* __restrict__ dst, int nheads, int total) {
  int tid = blockIdx.x * 256 + threadIdx.x;
  if (tid >= total) return;
  int ppr = nheads * 64;                      // pairs per row
  int s   = tid / ppr;
  int r   = tid % ppr;
  int p   = r & 63;
  int h   = r >> 6;
  size_t base = (size_t)s * (nheads * 128) + h * 128 + p * 2;
  float x1 = src[base], x2 = src[base + 1];
  float c  = cosb[s * 64 + p];
  float sn = sinb[s * 64 + p];
  dst[base]     = (__bf16)(x1 * c - x2 * sn);
  dst[base + 1] = (__bf16)(x1 * sn + x2 * c);
}

// ---------------------------------------------------------------------------
// 5) Flash-style causal attention. One wave per (q-tile of 16, head).
//    qb: bf16 [S x 4096] (post-RoPE)   -> A operand, row-major loads
//    kb: bf16 [S x 1024] (post-RoPE)   -> B operand for Q*K^T, row-major loads
//    vbp: packed B fragments per kv-head: [kvh][dt(8)][jt(SEQ/32)][lane][16]
//    attnb: bf16 [S x 4096] output
// ---------------------------------------------------------------------------
__global__ __launch_bounds__(32) void attn_kernel(
    const __bf16* __restrict__ qb,
    const __bf16* __restrict__ kb,
    const __bf16* __restrict__ vbp,
    __bf16* __restrict__ attnb) {
  __shared__ float  P[16][32];
  __shared__ __bf16 Pb[16][32];
  __shared__ float  rowmax[16], rowsum[16], rowscale[16];

  int lane = threadIdx.x;
  int it   = blockIdx.x;                      // q tile index (0..127)
  int h    = blockIdx.y;                      // head (0..31)
  int kvh  = h >> 2;                          // GQA: 4 q-heads per kv-head
  int l16  = lane & 15;
  int kofs = (lane >> 4) << 3;                // 0 or 8

  // Q fragments for all four K=32 chunks of HEAD_DIM=128
  bf16x16 qf[4];
  const __bf16* qrow = qb + (size_t)(it * 16 + l16) * QDIM + h * 128 + kofs;
#pragma unroll
  for (int kc = 0; kc < 4; ++kc) {
    bf16x8 a0 = gload8(qrow + kc * 32);
    bf16x8 a1 = gload8(qrow + kc * 32 + 16);
#pragma unroll
    for (int i = 0; i < 8; ++i) { qf[kc][i] = a0[i]; qf[kc][i + 8] = a1[i]; }
  }

  floatx8 o[8];
#pragma unroll
  for (int d = 0; d < 8; ++d) o[d] = (floatx8){};

  if (lane < 16) { rowmax[lane] = -1e30f; rowsum[lane] = 0.0f; }
  __syncthreads();

  const __bf16* vbh = vbp + (size_t)kvh * (SEQ * 128);

  int qlast  = it * 16 + 15;
  int jt_max = qlast >> 5;                    // 32-key tiles, inclusive
  for (int jt = 0; jt <= jt_max; ++jt) {
    // ---- scores for two 16-key sub-tiles -> LDS P[16][32] ----
#pragma unroll
    for (int n2 = 0; n2 < 2; ++n2) {
      int kbase = jt * 32 + n2 * 16;
      floatx8 s = {};
      const __bf16* krow =
          kb + (size_t)(kbase + l16) * KVDIM + kvh * 128 + kofs;
#pragma unroll
      for (int kc = 0; kc < 4; ++kc) {
        bf16x8 b0 = gload8(krow + kc * 32);
        bf16x8 b1 = gload8(krow + kc * 32 + 16);
        bf16x16 bv;
#pragma unroll
        for (int i = 0; i < 8; ++i) { bv[i] = b0[i]; bv[i + 8] = b1[i]; }
        s = __builtin_amdgcn_wmma_f32_16x16x32_bf16(
            false, qf[kc], false, bv, (short)0, s, false, false);
      }
#pragma unroll
      for (int r = 0; r < 8; ++r) {
        int m    = r + kofs;
        int qi   = it * 16 + m;
        int key  = kbase + l16;
        float v  = s[r] * SCALE_F;
        if (key > qi) v = -1e30f;             // causal mask
        P[m][n2 * 16 + l16] = v;
      }
    }
    __syncthreads();

    // ---- online softmax (lanes 0..15, one row each) ----
    if (lane < 16) {
      int m = lane;
      float mx = rowmax[m];
      for (int t = 0; t < 32; ++t) mx = fmaxf(mx, P[m][t]);
      float alpha = __expf(rowmax[m] - mx);
      float sum = 0.0f;
      for (int t = 0; t < 32; ++t) {
        float e = __expf(P[m][t] - mx);
        Pb[m][t] = (__bf16)e;
        sum += e;
      }
      rowsum[m]   = rowsum[m] * alpha + sum;
      rowscale[m] = alpha;
      rowmax[m]   = mx;
    }
    __syncthreads();

    float sc[8];
#pragma unroll
    for (int r = 0; r < 8; ++r) sc[r] = rowscale[r + kofs];

    // P as A fragment (16 rows x 32 keys)
    bf16x16 pf;
    {
      const __bf16* prow = &Pb[l16][kofs];
#pragma unroll
      for (int i = 0; i < 8; ++i) { pf[i] = prow[i]; pf[i + 8] = prow[i + 16]; }
    }

    // ---- O += P * V  (8 d-tiles of 16, K=32 keys) ----
#pragma unroll
    for (int dt = 0; dt < 8; ++dt) {
#pragma unroll
      for (int r = 0; r < 8; ++r) o[dt][r] *= sc[r];
      bf16x16 bv =
          gload16(vbh + (((size_t)dt * (SEQ / 32) + jt) * 32 + lane) * 16);
      o[dt] = __builtin_amdgcn_wmma_f32_16x16x32_bf16(
          false, pf, false, bv, (short)0, o[dt], false, false);
    }
    __syncthreads();                          // P/Pb reused next iteration
  }

  // ---- normalize and store bf16 row-major ----
  float inv[8];
#pragma unroll
  for (int r = 0; r < 8; ++r) inv[r] = 1.0f / rowsum[r + kofs];
#pragma unroll
  for (int dt = 0; dt < 8; ++dt)
#pragma unroll
    for (int r = 0; r < 8; ++r) {
      int m = r + kofs;
      attnb[(size_t)(it * 16 + m) * QDIM + h * 128 + dt * 16 + l16] =
          (__bf16)(o[dt][r] * inv[r]);
    }
}

// ---------------------------------------------------------------------------
// Host-side launcher
// ---------------------------------------------------------------------------
extern "C" void kernel_launch(void* const* d_in, const int* in_sizes, int n_in,
                              void* d_out, int out_size, void* d_ws,
                              size_t ws_size, hipStream_t stream) {
  const float* x  = (const float*)d_in[0];
  const float* fc = (const float*)d_in[1];
  const float* fs = (const float*)d_in[2];
  const float* wq = (const float*)d_in[3];
  const float* wk = (const float*)d_in[4];
  const float* wv = (const float*)d_in[5];
  const float* wo = (const float*)d_in[6];
  float* out = (float*)d_out;

  char* ws = (char*)d_ws;
  size_t off = 0;
  auto carve = [&](size_t bytes) -> void* {
    void* p = ws + off;
    off = (off + bytes + 255) & ~(size_t)255;
    return p;
  };

  __bf16* xb    = (__bf16*)carve((size_t)SEQ * DIM * 2);
  __bf16* wqp   = (__bf16*)carve((size_t)DIM * QDIM * 2);
  __bf16* wkp   = (__bf16*)carve((size_t)DIM * KVDIM * 2);
  __bf16* wvp   = (__bf16*)carve((size_t)DIM * KVDIM * 2);
  __bf16* wop   = (__bf16*)carve((size_t)QDIM * DIM * 2);
  float*  qf32  = (float*) carve((size_t)SEQ * QDIM * 4);
  float*  kf32  = (float*) carve((size_t)SEQ * KVDIM * 4);
  float*  vf32  = (float*) carve((size_t)SEQ * KVDIM * 4);
  __bf16* qbb   = (__bf16*)carve((size_t)SEQ * QDIM * 2);
  __bf16* kbb   = (__bf16*)carve((size_t)SEQ * KVDIM * 2);
  __bf16* vbp   = (__bf16*)carve((size_t)SEQ * KVDIM * 2);
  __bf16* attnb = (__bf16*)carve((size_t)SEQ * QDIM * 2);

  // 1) x -> bf16
  {
    int n = SEQ * DIM;
    convert_bf16_kernel<<<(n + 255) / 256, 256, 0, stream>>>(x, xb, n);
  }

  // 2) pack weights into B-fragment layout
  {
    int n = DIM * QDIM;
    pack_b_kernel<<<(n + 255) / 256, 256, 0, stream>>>(
        wq, QDIM, DIM / 32, QDIM / 16, wqp, n);
    pack_b_kernel<<<(n + 255) / 256, 256, 0, stream>>>(
        wo, DIM, QDIM / 32, DIM / 16, wop, n);
    int nk = DIM * KVDIM;
    pack_b_kernel<<<(nk + 255) / 256, 256, 0, stream>>>(
        wk, KVDIM, DIM / 32, KVDIM / 16, wkp, nk);
    pack_b_kernel<<<(nk + 255) / 256, 256, 0, stream>>>(
        wv, KVDIM, DIM / 32, KVDIM / 16, wvp, nk);
  }

  // 3) QKV projections (register-blocked WMMA GEMMs)
  {
    int wavesQ = (SEQ / 64) * (QDIM / 32);    // 32 * 128 = 4096
    gemm_wmma_bf16_kernel<<<wavesQ / 8, 256, 0, stream>>>(
        xb, DIM, wqp, qf32, QDIM, QDIM / 32, DIM / 32);
    int wavesK = (SEQ / 64) * (KVDIM / 32);   // 32 * 32 = 1024
    gemm_wmma_bf16_kernel<<<wavesK / 8, 256, 0, stream>>>(
        xb, DIM, wkp, kf32, KVDIM, KVDIM / 32, DIM / 32);
    gemm_wmma_bf16_kernel<<<wavesK / 8, 256, 0, stream>>>(
        xb, DIM, wvp, vf32, KVDIM, KVDIM / 32, DIM / 32);
  }

  // 4) RoPE on Q and K (f32 -> bf16)
  {
    int nq = SEQ * NHEADS * 64;
    rope_kernel<<<(nq + 255) / 256, 256, 0, stream>>>(
        qf32, fc, fs, qbb, NHEADS, nq);
    int nk = SEQ * NKVHEADS * 64;
    rope_kernel<<<(nk + 255) / 256, 256, 0, stream>>>(
        kf32, fc, fs, kbb, NKVHEADS, nk);
  }

  // 5) Pack V per kv-head into B-fragment layout: [kvh][dt][jt][lane][16]
  for (int kvh = 0; kvh < NKVHEADS; ++kvh) {
    int n = SEQ * HEADDIM;                    // 262144
    pack_b_kernel<<<(n + 255) / 256, 256, 0, stream>>>(
        vf32 + kvh * HEADDIM, KVDIM, SEQ / 32, HEADDIM / 16,
        vbp + (size_t)kvh * SEQ * HEADDIM, n);
  }

  // 6) causal flash attention
  {
    dim3 grid(SEQ / 16, NHEADS);
    attn_kernel<<<grid, 32, 0, stream>>>(qbb, kbb, vbp, attnb);
  }

  // 7) output projection
  {
    int waves = (SEQ / 64) * (DIM / 32);      // 4096
    gemm_wmma_bf16_kernel<<<waves / 8, 256, 0, stream>>>(
        attnb, QDIM, wop, out, DIM, DIM / 32, QDIM / 32);
  }
}
